// RNNModel_70961449664765
// MI455X (gfx1250) — compile-verified, matrix operand order checked by reading, not run on hardware
//
#include <hip/hip_runtime.h>
#include <math.h>

// ---------------------------------------------------------------------------
// Types for CDNA5 WMMA (wave32): v16bf A/B operands, v8f f32 accumulator.
// ---------------------------------------------------------------------------
typedef __attribute__((ext_vector_type(16))) __bf16         v16bf;
typedef __attribute__((ext_vector_type(16))) unsigned short v16us;
typedef __attribute__((ext_vector_type(8)))  unsigned int   v8u;
typedef __attribute__((ext_vector_type(8)))  float          v8f;

#define HID   1024
#define BATCH 128
#define SEQT  256
#define EMB   256
#define VOC   256

// RNE float -> bf16
__device__ __forceinline__ unsigned short f2bf(float f) {
    unsigned int u = __builtin_bit_cast(unsigned int, f);
    unsigned int r = (u + 0x7FFFu + ((u >> 16) & 1u)) >> 16;
    return (unsigned short)r;
}

// K offset pattern for 16-bit WMMA A (and mirrored B) fragments:
// VGPR j holds K pair (k0,k0+1); lanes 0-15 cover K {0..7,16..23},
// lanes 16-31 cover K {8..15,24..31}.
__device__ __forceinline__ int frag_k0(int j, int half) {
    return (j < 4) ? (2 * j + 8 * half) : (2 * (j - 4) + 8 * half + 16);
}

// Assemble a 16x32 bf16 A fragment from row-major data.
// p = row base for this lane's M row, advanced to the K-tile start.
__device__ __forceinline__ v16bf load_a_frag(const unsigned short* __restrict__ p, int half) {
    v8u r;
#pragma unroll
    for (int j = 0; j < 8; ++j) {
        r[j] = *(const unsigned int*)(p + frag_k0(j, half));
    }
    return __builtin_bit_cast(v16bf, r);
}

// B fragment pre-packed in per-lane order: one coalesced 32B load.
__device__ __forceinline__ v16bf load_b_frag(const unsigned short* __restrict__ p) {
    return __builtin_bit_cast(v16bf, *(const v16us*)p);
}

__device__ __forceinline__ v8f wmma_bf16(v16bf a, v16bf b, v8f c) {
    return __builtin_amdgcn_wmma_f32_16x16x32_bf16(false, a, false, b, (short)0, c, false, false);
}

__device__ __forceinline__ float sigmoidf_(float x) {
    return 1.0f / (1.0f + __expf(-x));
}

// ---------------------------------------------------------------------------
// Prep kernels
// ---------------------------------------------------------------------------

// xe[t][b][e] = bf16(emb[x[b][t]][e])
__global__ __launch_bounds__(256) void embed_kernel(const int* __restrict__ x,
                                                    const float* __restrict__ emb,
                                                    unsigned short* __restrict__ xe) {
    int idx = blockIdx.x * 256 + threadIdx.x;           // t*B*E + b*E + e
    int e = idx & (EMB - 1);
    int b = (idx >> 8) & (BATCH - 1);
    int t = idx >> 15;
    xe[idx] = f2bf(emb[(size_t)x[b * SEQT + t] * EMB + e]);
}

// Pack row-major W [N][K] (used as B matrix B[k][n] = W[n][k]) into
// WMMA B-fragment order: out[((nt*Ktiles + kt)*32 + lane)*16 + 2j + s].
__global__ __launch_bounds__(32) void pack_b_kernel(const float* __restrict__ W,
                                                    unsigned short* __restrict__ out,
                                                    int K) {
    int Ktiles = K >> 5;
    int tile = blockIdx.x;            // nt*Ktiles + kt
    int nt = tile / Ktiles;
    int kt = tile - nt * Ktiles;
    int lane = threadIdx.x;
    int nl = lane & 15, half = lane >> 4;
    const float* wrow = W + (size_t)(nt * 16 + nl) * K + kt * 32;
    unsigned short* o = out + ((size_t)tile * 32 + lane) * 16;
#pragma unroll
    for (int j = 0; j < 8; ++j) {
        int k0 = frag_k0(j, half);
        o[2 * j]     = f2bf(wrow[k0]);
        o[2 * j + 1] = f2bf(wrow[k0 + 1]);
    }
}

__global__ __launch_bounds__(256) void f32_to_bf16_kernel(const float* __restrict__ src,
                                                          unsigned short* __restrict__ dst,
                                                          int n) {
    int i = blockIdx.x * 256 + threadIdx.x;
    if (i < n) dst[i] = f2bf(src[i]);
}

// ---------------------------------------------------------------------------
// Fused GRU time step. grid = (64, 8): 64 column tiles of H, 8 row tiles of B.
// One wave per (mt,nt). Four accumulators: r,z fused (input+hidden K),
// i_n (input K only), h_n (hidden K only)  -- because n = tanh(i_n + r*h_n).
// ---------------------------------------------------------------------------
__global__ __launch_bounds__(32) void gru_step_kernel(
    const unsigned short* __restrict__ x_t,   // [B][Kx] bf16
    int Ktx,                                  // Kx / 32
    const unsigned short* __restrict__ Wih_p, // packed [192][Ktx][32][16]
    const unsigned short* __restrict__ h_bf,  // [B][H] bf16 (prev hidden)
    const unsigned short* __restrict__ Whh_p, // packed [192][32][32][16]
    const float* __restrict__ b_ih,
    const float* __restrict__ b_hh,
    const float* __restrict__ h_prev,         // [B][H] f32
    float* __restrict__ h_out_f32,            // [B][H] f32
    unsigned short* __restrict__ h_out_bf)    // [B][H] bf16
{
    const int nt = blockIdx.x;                // 0..63
    const int mt = blockIdx.y;                // 0..7
    const int lane = threadIdx.x;
    const int nl = lane & 15, half = lane >> 4;
    const int m = lane & 15;                  // A-fragment row
    const int Kx = Ktx << 5;

    v8f acc_r = {}, acc_z = {}, acc_in = {}, acc_hn = {};

    // ----- input-side GEMM: gates r,z accumulate; i_n separate -----
    {
        const unsigned short* arow = x_t + (size_t)(mt * 16 + m) * Kx;
        const unsigned short* br = Wih_p + (size_t)(nt)       * Ktx * 512;
        const unsigned short* bz = Wih_p + (size_t)(nt + 64)  * Ktx * 512;
        const unsigned short* bn = Wih_p + (size_t)(nt + 128) * Ktx * 512;
        for (int kt = 0; kt < Ktx; ++kt) {
            v16bf a = load_a_frag(arow + kt * 32, half);
            size_t off = ((size_t)kt * 32 + lane) * 16;
            acc_r  = wmma_bf16(a, load_b_frag(br + off), acc_r);
            acc_z  = wmma_bf16(a, load_b_frag(bz + off), acc_z);
            acc_in = wmma_bf16(a, load_b_frag(bn + off), acc_in);
        }
    }

    // ----- hidden-side GEMM: gates r,z accumulate; h_n separate -----
    {
        const unsigned short* arow = h_bf + (size_t)(mt * 16 + m) * HID;
        const unsigned short* br = Whh_p + (size_t)(nt)       * 32 * 512;
        const unsigned short* bz = Whh_p + (size_t)(nt + 64)  * 32 * 512;
        const unsigned short* bn = Whh_p + (size_t)(nt + 128) * 32 * 512;
        for (int kt = 0; kt < 32; ++kt) {
            v16bf a = load_a_frag(arow + kt * 32, half);
            size_t off = ((size_t)kt * 32 + lane) * 16;
            acc_r  = wmma_bf16(a, load_b_frag(br + off), acc_r);
            acc_z  = wmma_bf16(a, load_b_frag(bz + off), acc_z);
            acc_hn = wmma_bf16(a, load_b_frag(bn + off), acc_hn);
        }
    }

    // ----- element-wise GRU update in registers -----
    const int col = nt * 16 + nl;
    const float bir = b_ih[col],        bhr = b_hh[col];
    const float biz = b_ih[HID + col],  bhz = b_hh[HID + col];
    const float bin = b_ih[2*HID + col], bhn = b_hh[2*HID + col];
#pragma unroll
    for (int i = 0; i < 8; ++i) {
        int mb = mt * 16 + i + 8 * half;      // C/D layout: lanes16-31 -> M+8
        float r  = sigmoidf_(acc_r[i] + bir + bhr);
        float z  = sigmoidf_(acc_z[i] + biz + bhz);
        float nn = tanhf(acc_in[i] + bin + r * (acc_hn[i] + bhn));
        float hp = h_prev[(size_t)mb * HID + col];
        float hn = (1.0f - z) * nn + z * hp;
        h_out_f32[(size_t)mb * HID + col] = hn;
        h_out_bf[(size_t)mb * HID + col]  = f2bf(hn);
    }
}

// ---------------------------------------------------------------------------
// Final FC: logits[B][VOC] = h1 @ fc_W^T + fc_b.   grid = (16, 8).
// ---------------------------------------------------------------------------
__global__ __launch_bounds__(32) void fc_kernel(const unsigned short* __restrict__ h_bf,
                                                const unsigned short* __restrict__ W_p,
                                                const float* __restrict__ fc_b,
                                                float* __restrict__ out) {
    const int nt = blockIdx.x;                // 0..15
    const int mt = blockIdx.y;                // 0..7
    const int lane = threadIdx.x;
    const int nl = lane & 15, half = lane >> 4;
    const int m = lane & 15;

    v8f acc = {};
    const unsigned short* arow = h_bf + (size_t)(mt * 16 + m) * HID;
    const unsigned short* bp = W_p + (size_t)nt * 32 * 512;
    for (int kt = 0; kt < 32; ++kt) {
        v16bf a = load_a_frag(arow + kt * 32, half);
        acc = wmma_bf16(a, load_b_frag(bp + ((size_t)kt * 32 + lane) * 16), acc);
    }
    const int col = nt * 16 + nl;
    const float bias = fc_b[col];
#pragma unroll
    for (int i = 0; i < 8; ++i) {
        int mb = mt * 16 + i + 8 * half;
        out[(size_t)mb * VOC + col] = acc[i] + bias;
    }
}

// ---------------------------------------------------------------------------
// Host orchestration
// ---------------------------------------------------------------------------
extern "C" void kernel_launch(void* const* d_in, const int* in_sizes, int n_in,
                              void* d_out, int out_size, void* d_ws, size_t ws_size,
                              hipStream_t stream) {
    const int*   x     = (const int*)  d_in[0];
    const float* h0    = (const float*)d_in[1];
    const float* emb   = (const float*)d_in[2];
    const float* W_ih0 = (const float*)d_in[3];
    const float* W_hh0 = (const float*)d_in[4];
    const float* b_ih0 = (const float*)d_in[5];
    const float* b_hh0 = (const float*)d_in[6];
    const float* W_ih1 = (const float*)d_in[7];
    const float* W_hh1 = (const float*)d_in[8];
    const float* b_ih1 = (const float*)d_in[9];
    const float* b_hh1 = (const float*)d_in[10];
    const float* fc_W  = (const float*)d_in[11];
    const float* fc_b  = (const float*)d_in[12];
    float* out = (float*)d_out;

    // workspace carve-out (all offsets 256B aligned)
    char* ws = (char*)d_ws;
    size_t off = 0;
    auto take = [&](size_t bytes) -> void* {
        void* p = ws + off;
        off += (bytes + 255) & ~(size_t)255;
        return p;
    };
    unsigned short* xe     = (unsigned short*)take((size_t)SEQT * BATCH * EMB * 2);
    unsigned short* y0     = (unsigned short*)take((size_t)SEQT * BATCH * HID * 2);
    unsigned short* Wih0p  = (unsigned short*)take((size_t)192 * 8  * 512 * 2);
    unsigned short* Whh0p  = (unsigned short*)take((size_t)192 * 32 * 512 * 2);
    unsigned short* Wih1p  = (unsigned short*)take((size_t)192 * 32 * 512 * 2);
    unsigned short* Whh1p  = (unsigned short*)take((size_t)192 * 32 * 512 * 2);
    unsigned short* fcWp   = (unsigned short*)take((size_t)16  * 32 * 512 * 2);
    unsigned short* h0bf0  = (unsigned short*)take((size_t)BATCH * HID * 2);
    unsigned short* h0bf1  = (unsigned short*)take((size_t)BATCH * HID * 2);
    float* hbuf0[2] = { (float*)take((size_t)BATCH * HID * 4),
                        (float*)take((size_t)BATCH * HID * 4) };
    float* hbuf1[2] = { (float*)take((size_t)BATCH * HID * 4),
                        (float*)take((size_t)BATCH * HID * 4) };
    unsigned short* hbf1[2] = { (unsigned short*)take((size_t)BATCH * HID * 2),
                                (unsigned short*)take((size_t)BATCH * HID * 2) };
    (void)ws_size; (void)in_sizes; (void)n_in; (void)out_size;

    // ---- prep: embedding, weight repack, h0 conversion ----
    embed_kernel<<<(SEQT * BATCH * EMB) / 256, 256, 0, stream>>>(x, emb, xe);
    pack_b_kernel<<<192 * 8,  32, 0, stream>>>(W_ih0, Wih0p, EMB);
    pack_b_kernel<<<192 * 32, 32, 0, stream>>>(W_hh0, Whh0p, HID);
    pack_b_kernel<<<192 * 32, 32, 0, stream>>>(W_ih1, Wih1p, HID);
    pack_b_kernel<<<192 * 32, 32, 0, stream>>>(W_hh1, Whh1p, HID);
    pack_b_kernel<<<16  * 32, 32, 0, stream>>>(fc_W,  fcWp,  HID);
    f32_to_bf16_kernel<<<(BATCH * HID + 255) / 256, 256, 0, stream>>>(h0, h0bf0, BATCH * HID);
    f32_to_bf16_kernel<<<(BATCH * HID + 255) / 256, 256, 0, stream>>>(h0 + (size_t)BATCH * HID,
                                                                     h0bf1, BATCH * HID);

    const dim3 gridStep(64, 8), blockStep(32);
    float* hlast0 = out + (size_t)BATCH * VOC;                 // [B,H] layer0 final
    float* hlast1 = hlast0 + (size_t)BATCH * HID;              // [B,H] layer1 final

    // ---- layer 0 scan: x-side K=256, hidden K=1024; y0[t] = new hidden (bf16) ----
    {
        const float* hprev = h0;
        const unsigned short* hbf_in = h0bf0;
        for (int t = 0; t < SEQT; ++t) {
            float* hout = (t == SEQT - 1) ? hlast0 : hbuf0[t & 1];
            unsigned short* hbfout = y0 + (size_t)t * BATCH * HID;
            gru_step_kernel<<<gridStep, blockStep, 0, stream>>>(
                xe + (size_t)t * BATCH * EMB, 8, Wih0p,
                hbf_in, Whh0p, b_ih0, b_hh0, hprev, hout, hbfout);
            hprev = hout;
            hbf_in = hbfout;
        }
    }

    // ---- layer 1 scan: input = y0[t] (K=1024) ----
    const unsigned short* h1_final = nullptr;
    {
        const float* hprev = h0 + (size_t)BATCH * HID;
        const unsigned short* hbf_in = h0bf1;
        for (int t = 0; t < SEQT; ++t) {
            float* hout = (t == SEQT - 1) ? hlast1 : hbuf1[t & 1];
            unsigned short* hbfout = hbf1[t & 1];
            gru_step_kernel<<<gridStep, blockStep, 0, stream>>>(
                y0 + (size_t)t * BATCH * HID, 32, Wih1p,
                hbf_in, Whh1p, b_ih1, b_hh1, hprev, hout, hbfout);
            hprev = hout;
            hbf_in = hbfout;
        }
        h1_final = hbf_in;
    }

    // ---- final FC: logits = h1 @ fc_W^T + fc_b ----
    fc_kernel<<<dim3(16, 8), 32, 0, stream>>>(h1_final, fcWp, fc_b, out);
}
